// Qwen35Attention_51058571215202
// MI455X (gfx1250) — matrix operand order repaired and checked by reading, not compiled, not run on hardware
//
#include <hip/hip_runtime.h>
#include <hip/hip_bf16.h>
#include <stdint.h>

#define B_ 2
#define S_ 2048
#define H_ 2048
#define NH_ 16
#define NKV_ 4
#define D_ 128
#define GROUPS_ (NH_ / NKV_)
#define SCALE_ 0.08838834764831845f   // 128^-0.5
#define EPS_ 1e-6f

typedef __attribute__((ext_vector_type(16))) __bf16 v16bf;
typedef __attribute__((ext_vector_type(8)))  float  v8f;
typedef __attribute__((ext_vector_type(4)))  int    v4i;

union Frag16 { v16bf v; uint32_t u[8]; };
union FragC  { v8f  v; float    f[8]; };

#if defined(__has_builtin)
#if __has_builtin(__builtin_amdgcn_global_load_async_to_lds_b128)
#define HAVE_ASYNC_LDS 1
#endif
#endif
#ifndef HAVE_ASYNC_LDS
#define HAVE_ASYNC_LDS 0
#endif

#define AS1 __attribute__((address_space(1)))
#define AS3 __attribute__((address_space(3)))
typedef AS1 v4i* gptr_v4i;
typedef AS3 v4i* lptr_v4i;

__device__ __forceinline__ uint16_t f2bf(float x) {
  uint32_t u = __float_as_uint(x);
  u += 0x7fffu + ((u >> 16) & 1u);          // round-to-nearest-even
  return (uint16_t)(u >> 16);
}
__device__ __forceinline__ uint32_t f2bf2(float lo, float hi) {
  return (uint32_t)f2bf(lo) | ((uint32_t)f2bf(hi) << 16);
}

// ---------------------------------------------------------------------------
// Tiled GEMM: C[M,N] = X[M,K] * W[K,N], fp32 in/out, bf16 WMMA compute.
// Block tile 64x128, K-step 32. 8 waves in 4(M)x2(N) grid; each wave 16x64.
// LDS: A row-major [m][k] (K-pairs contiguous -> A-frag b128 loads),
//      B transposed [n][k]  (K-pairs contiguous -> B-frag b128 loads).
// ---------------------------------------------------------------------------
__global__ __launch_bounds__(256) void gemm_xw_bf16(
    const float* __restrict__ X, const float* __restrict__ W,
    float* __restrict__ C, int M, int N, int K) {
  __shared__ uint16_t sA[64 * 32];
  __shared__ uint16_t sB[128 * 32];
  uint32_t* sA32 = (uint32_t*)sA;

  const int tid  = threadIdx.x;
  const int lane = tid & 31;
  const int wv   = tid >> 5;
  const int wm   = (wv & 3) * 16;
  const int wn   = (wv >> 2) * 64;
  const int m0   = blockIdx.x * 64;
  const int n0   = blockIdx.y * 128;

  FragC acc[4];
#pragma unroll
  for (int t = 0; t < 4; ++t)
#pragma unroll
    for (int r = 0; r < 8; ++r) acc[t].f[r] = 0.f;

  for (int k0 = 0; k0 < K; k0 += 32) {
    // prefetch next K-step tiles into L2 while this step is processed
    if (k0 + 32 < K) {
      __builtin_prefetch(X + (size_t)(m0 + (tid >> 2)) * K + k0 + 32, 0, 0);
      __builtin_prefetch(W + (size_t)(k0 + 32 + (tid >> 3)) * N + n0 + (tid & 7) * 16, 0, 0);
    }
    __syncthreads();
    // A tile 64x32: 1024 bf16-pairs, 4 per thread
#pragma unroll
    for (int i = 0; i < 4; ++i) {
      int p = tid * 4 + i;                    // 0..1023
      int m = p >> 4, kp = (p & 15) * 2;
      const float* src = X + (size_t)(m0 + m) * K + k0 + kp;
      sA32[m * 16 + (p & 15)] = f2bf2(src[0], src[1]);
    }
    // B tile 32x128 -> transposed sB[n][k]
    {
      int k  = tid >> 3;
      int nb = (tid & 7) * 16;
      const float* src = W + (size_t)(k0 + k) * N + n0 + nb;
#pragma unroll
      for (int j = 0; j < 16; ++j)
        sB[(nb + j) * 32 + k] = f2bf(src[j]);
    }
    __syncthreads();

    Frag16 afr;
#pragma unroll
    for (int v = 0; v < 8; ++v) {
      int kb = (v >> 2) * 16 + ((lane >> 4) << 3) + ((v & 3) << 1);
      int m  = wm + (lane & 15);
      afr.u[v] = *(const uint32_t*)&sA[m * 32 + kb];
    }
#pragma unroll
    for (int t = 0; t < 4; ++t) {
      Frag16 bfr;
      int n = wn + t * 16 + (lane & 15);
#pragma unroll
      for (int v = 0; v < 8; ++v) {
        int kb = (v >> 2) * 16 + ((lane >> 4) << 3) + ((v & 3) << 1);
        bfr.u[v] = *(const uint32_t*)&sB[n * 32 + kb];
      }
      acc[t].v = __builtin_amdgcn_wmma_f32_16x16x32_bf16(
          false, afr.v, false, bfr.v, (short)0, acc[t].v, false, false);
    }
  }

#pragma unroll
  for (int t = 0; t < 4; ++t) {
    int col = n0 + wn + t * 16 + (lane & 15);
#pragma unroll
    for (int r = 0; r < 8; ++r) {
      int row = m0 + wm + r + ((lane >> 4) << 3);
      C[(size_t)row * N + col] = acc[t].f[r];
    }
  }
}

// ---------------------------------------------------------------------------
// Per-head RMSNorm + RoPE, fp32 -> bf16, layout [B, nheads, S, D].
// One wave per (b,s,h) row; lane handles 4 contiguous d; rotate_half pairing
// via shfl_xor(.,16) (lane<16 holds d<64, lane+16 holds d+64).
// head_stride = 2*D for q (read first D of each head slot), D for k.
// ---------------------------------------------------------------------------
__global__ __launch_bounds__(256) void prep_norm_rope(
    const float* __restrict__ proj, const float* __restrict__ wnorm,
    const float* __restrict__ cosT, const float* __restrict__ sinT,
    uint16_t* __restrict__ outb, int nheads, int head_stride) {
  const int tid = threadIdx.x, lane = tid & 31, wv = tid >> 5;
  const int gw = blockIdx.x * 8 + wv;        // row over B*S*nheads
  const int h = gw % nheads;
  const int s = (gw / nheads) % S_;
  const int b = gw / (nheads * S_);
  const int rowW = nheads * head_stride;
  const float* src = proj + (size_t)(b * S_ + s) * rowW + h * head_stride;
  const int d0 = lane * 4;

  float4 x = *(const float4*)(src + d0);
  float ss = x.x * x.x + x.y * x.y + x.z * x.z + x.w * x.w;
  ss += __shfl_xor(ss, 1);  ss += __shfl_xor(ss, 2);
  ss += __shfl_xor(ss, 4);  ss += __shfl_xor(ss, 8);
  ss += __shfl_xor(ss, 16);
  float rr = rsqrtf(ss * (1.f / D_) + EPS_);
  float4 w4 = *(const float4*)(wnorm + d0);
  float n0 = x.x * rr * w4.x, n1 = x.y * rr * w4.y;
  float n2 = x.z * rr * w4.z, n3 = x.w * rr * w4.w;
  float p0 = __shfl_xor(n0, 16), p1 = __shfl_xor(n1, 16);
  float p2 = __shfl_xor(n2, 16), p3 = __shfl_xor(n3, 16);
  float sgn = (lane < 16) ? -1.f : 1.f;
  float4 c4 = *(const float4*)(cosT + (size_t)s * D_ + d0);
  float4 s4 = *(const float4*)(sinT + (size_t)s * D_ + d0);
  float o0 = n0 * c4.x + sgn * p0 * s4.x;
  float o1 = n1 * c4.y + sgn * p1 * s4.y;
  float o2 = n2 * c4.z + sgn * p2 * s4.z;
  float o3 = n3 * c4.w + sgn * p3 * s4.w;
  size_t oidx = ((size_t)(b * nheads + h) * S_ + s) * D_ + d0;
  uint32_t* o32 = (uint32_t*)(outb + oidx);
  o32[0] = f2bf2(o0, o1);
  o32[1] = f2bf2(o2, o3);
}

// V: fp32 [B,S,NKV,D] -> bf16 [B,NKV,S,D]
__global__ __launch_bounds__(256) void prep_v(
    const float* __restrict__ vproj, uint16_t* __restrict__ vb) {
  const int i0 = (blockIdx.x * 256 + threadIdx.x) * 4;
  const int d  = i0 % D_;
  const int kv = (i0 / D_) % NKV_;
  const int s  = (i0 / (D_ * NKV_)) % S_;
  const int b  = i0 / (D_ * NKV_ * S_);
  float4 x = *(const float4*)(vproj + i0);
  size_t o = ((size_t)(b * NKV_ + kv) * S_ + s) * D_ + d;
  uint32_t* o32 = (uint32_t*)(vb + o);
  o32[0] = f2bf2(x.x, x.y);
  o32[1] = f2bf2(x.z, x.w);
}

// ---------------------------------------------------------------------------
// Flash attention: block = (q-tile of 128, head h, batch b), 8 waves x 16 rows.
// K-tile is a contiguous 8KB block -> staged via async global->LDS copy
// (ASYNCcnt) when available; V transposed manually; online softmax wave-local;
// P -> LDS -> A-fragment for PV; gate fused in epilogue.
// ---------------------------------------------------------------------------
__global__ __launch_bounds__(256) void flash_attn(
    const uint16_t* __restrict__ qb, const uint16_t* __restrict__ kb,
    const uint16_t* __restrict__ vb, const float* __restrict__ qg,
    float* __restrict__ out) {
  __shared__ uint16_t sK[32 * 128];    // [key][d]
  __shared__ uint16_t sV[128 * 32];    // [d][key]
  __shared__ uint16_t sP[8 * 16 * 32]; // per-wave P 16x32
  uint32_t* sK32 = (uint32_t*)sK;

  const int tid = threadIdx.x, lane = tid & 31, wv = tid >> 5;
  const int qt = blockIdx.x, h = blockIdx.y, b = blockIdx.z;
  const int kvh = h / GROUPS_;
  const int qbase = qt * 128;

  // preload Q fragments (4 chunks of K=32)
  const uint32_t* q32 = (const uint32_t*)qb;
  const int qrowA = qbase + wv * 16 + (lane & 15);
  const size_t qrb = ((size_t)(b * NH_ + h) * S_ + qrowA) * D_;
  Frag16 qfr[4];
#pragma unroll
  for (int c = 0; c < 4; ++c)
#pragma unroll
    for (int v = 0; v < 8; ++v) {
      int kbv = (v >> 2) * 16 + ((lane >> 4) << 3) + ((v & 3) << 1);
      qfr[c].u[v] = q32[(qrb + c * 32 + kbv) >> 1];
    }

  FragC ofr[8];
#pragma unroll
  for (int t = 0; t < 8; ++t)
#pragma unroll
    for (int r = 0; r < 8; ++r) ofr[t].f[r] = 0.f;
  float Mrun[8], Lrun[8];
#pragma unroll
  for (int r = 0; r < 8; ++r) { Mrun[r] = -1e30f; Lrun[r] = 0.f; }

  const int nkt = (qbase + 128) / 32;
  const uint32_t* kb32 = (const uint32_t*)kb;
  const uint32_t* vb32 = (const uint32_t*)vb;
  const size_t kvb = (size_t)(b * NKV_ + kvh) * S_ * (D_ / 2);

  for (int kt = 0; kt < nkt; ++kt) {
    const int k0 = kt * 32;
    // prefetch next tile's K/V global lines (32B per thread covers 8KB each)
    if (kt + 1 < nkt) {
      __builtin_prefetch((const char*)(kb32 + kvb + (size_t)(k0 + 32) * 64) + tid * 32, 0, 0);
      __builtin_prefetch((const char*)(vb32 + kvb + (size_t)(k0 + 32) * 64) + tid * 32, 0, 0);
    }
    __syncthreads();
#if HAVE_ASYNC_LDS
    {   // K tile: contiguous 8KB global -> LDS, 2 x b128 per thread
      char* gsrc = (char*)(kb32 + kvb + (size_t)k0 * 64);
#pragma unroll
      for (int j = 0; j < 2; ++j) {
        int c = j * 256 + tid;                // 512 chunks of 16B
        __builtin_amdgcn_global_load_async_to_lds_b128(
            (gptr_v4i)(gsrc + c * 16),
            (lptr_v4i)((char*)sK + c * 16), 0, 0);
      }
    }
#else
#pragma unroll
    for (int j = 0; j < 8; ++j) {             // K tile: 2048 u32
      int e = j * 256 + tid;
      int key = e >> 6, dp = e & 63;
      sK32[key * 64 + dp] = kb32[kvb + (size_t)(k0 + key) * 64 + dp];
    }
#endif
#pragma unroll
    for (int j = 0; j < 8; ++j) {             // V tile, transposed store
      int e = j * 256 + tid;
      int key = e >> 6, dp = e & 63;
      uint32_t val = vb32[kvb + (size_t)(k0 + key) * 64 + dp];
      sV[(dp * 2) * 32 + key]     = (uint16_t)val;
      sV[(dp * 2 + 1) * 32 + key] = (uint16_t)(val >> 16);
    }
#if HAVE_ASYNC_LDS
    asm volatile("s_wait_asynccnt 0x0" ::: "memory");
#endif
    __syncthreads();

    // S = Q K^T  (two 16-key halves)
    FragC sfr[2];
#pragma unroll
    for (int hn = 0; hn < 2; ++hn) {
#pragma unroll
      for (int r = 0; r < 8; ++r) sfr[hn].f[r] = 0.f;
      int n = hn * 16 + (lane & 15);
#pragma unroll
      for (int c = 0; c < 4; ++c) {
        Frag16 bfr;
#pragma unroll
        for (int v = 0; v < 8; ++v) {
          int kbv = (v >> 2) * 16 + ((lane >> 4) << 3) + ((v & 3) << 1);
          bfr.u[v] = sK32[(n * 128 + c * 32 + kbv) >> 1];
        }
        sfr[hn].v = __builtin_amdgcn_wmma_f32_16x16x32_bf16(
            false, qfr[c].v, false, bfr.v, (short)0, sfr[hn].v, false, false);
      }
    }

    // scale + causal mask + online softmax (row lives in one 16-lane group)
#pragma unroll
    for (int r = 0; r < 8; ++r) {
      int m = r + ((lane >> 4) << 3);
      int qrow = qbase + wv * 16 + m;
      int key0 = k0 + (lane & 15);
      float v0 = sfr[0].f[r] * SCALE_ + ((key0 > qrow) ? -1e9f : 0.f);
      float v1 = sfr[1].f[r] * SCALE_ + ((key0 + 16 > qrow) ? -1e9f : 0.f);
      float mx = fmaxf(v0, v1);
      mx = fmaxf(mx, __shfl_xor(mx, 1));
      mx = fmaxf(mx, __shfl_xor(mx, 2));
      mx = fmaxf(mx, __shfl_xor(mx, 4));
      mx = fmaxf(mx, __shfl_xor(mx, 8));
      float Mn = fmaxf(Mrun[r], mx);
      float alpha = __expf(Mrun[r] - Mn);
      Mrun[r] = Mn;
      float p0 = __expf(v0 - Mn);
      float p1 = __expf(v1 - Mn);
      float ls = p0 + p1;
      ls += __shfl_xor(ls, 1);  ls += __shfl_xor(ls, 2);
      ls += __shfl_xor(ls, 4);  ls += __shfl_xor(ls, 8);
      Lrun[r] = Lrun[r] * alpha + ls;
#pragma unroll
      for (int t = 0; t < 8; ++t) ofr[t].f[r] *= alpha;
      sP[wv * 512 + m * 32 + (lane & 15)]      = f2bf(p0);
      sP[wv * 512 + m * 32 + 16 + (lane & 15)] = f2bf(p1);
    }
    asm volatile("s_wait_dscnt 0x0" ::: "memory"); // wave-local P store->load order

    // O += P V
    Frag16 pfr;
#pragma unroll
    for (int v = 0; v < 8; ++v) {
      int kbv = (v >> 2) * 16 + ((lane >> 4) << 3) + ((v & 3) << 1);
      pfr.u[v] = *(const uint32_t*)&sP[wv * 512 + (lane & 15) * 32 + kbv];
    }
#pragma unroll
    for (int t = 0; t < 8; ++t) {
      Frag16 bfr;
      int dc = t * 16 + (lane & 15);
#pragma unroll
      for (int v = 0; v < 8; ++v) {
        int kbv = (v >> 2) * 16 + ((lane >> 4) << 3) + ((v & 3) << 1);
        bfr.u[v] = *(const uint32_t*)&sV[dc * 32 + kbv];
      }
      ofr[t].v = __builtin_amdgcn_wmma_f32_16x16x32_bf16(
          false, pfr.v, false, bfr.v, (short)0, ofr[t].v, false, false);
    }
  }

  // epilogue: normalize + sigmoid(gate)
#pragma unroll
  for (int t = 0; t < 8; ++t) {
    int dc = t * 16 + (lane & 15);
#pragma unroll
    for (int r = 0; r < 8; ++r) {
      int m = r + ((lane >> 4) << 3);
      int qrow = qbase + wv * 16 + m;
      float val = ofr[t].f[r] / Lrun[r];
      float g = qg[(size_t)(b * S_ + qrow) * (NH_ * 2 * D_) + h * 2 * D_ + D_ + dc];
      out[(size_t)(b * S_ + qrow) * (NH_ * D_) + h * D_ + dc] =
          val * (1.f / (1.f + __expf(-g)));
    }
  }
}

// ---------------------------------------------------------------------------
extern "C" void kernel_launch(void* const* d_in, const int* in_sizes, int n_in,
                              void* d_out, int out_size, void* d_ws, size_t ws_size,
                              hipStream_t stream) {
  const float* hidden = (const float*)d_in[0];
  const float* Wq     = (const float*)d_in[1];
  const float* Wk     = (const float*)d_in[2];
  const float* Wv     = (const float*)d_in[3];
  const float* qnw    = (const float*)d_in[4];
  const float* knw    = (const float*)d_in[5];
  const float* cosT   = (const float*)d_in[6];
  const float* sinT   = (const float*)d_in[7];
  // d_in[8] attention_mask: causal, applied analytically -> never read
  float* outp = (float*)d_out;

  char* ws = (char*)d_ws;
  size_t off = 0;
  auto carve = [&](size_t bytes) {
    char* p = ws + off;
    off += (bytes + 255) & ~(size_t)255;
    return p;
  };
  float*    qgm   = (float*)   carve((size_t)B_ * S_ * NH_ * 2 * D_ * 4);
  float*    kproj = (float*)   carve((size_t)B_ * S_ * NKV_ * D_ * 4);
  float*    vproj = (float*)   carve((size_t)B_ * S_ * NKV_ * D_ * 4);
  uint16_t* qbf   = (uint16_t*)carve((size_t)B_ * NH_ * S_ * D_ * 2);
  uint16_t* kbf   = (uint16_t*)carve((size_t)B_ * NKV_ * S_ * D_ * 2);
  uint16_t* vbf   = (uint16_t*)carve((size_t)B_ * NKV_ * S_ * D_ * 2);

  const int M = B_ * S_;
  gemm_xw_bf16<<<dim3(M / 64, (NH_ * 2 * D_) / 128), 256, 0, stream>>>(
      hidden, Wq, qgm, M, NH_ * 2 * D_, H_);
  gemm_xw_bf16<<<dim3(M / 64, (NKV_ * D_) / 128), 256, 0, stream>>>(
      hidden, Wk, kproj, M, NKV_ * D_, H_);
  gemm_xw_bf16<<<dim3(M / 64, (NKV_ * D_) / 128), 256, 0, stream>>>(
      hidden, Wv, vproj, M, NKV_ * D_, H_);
  prep_norm_rope<<<(M * NH_) / 8, 256, 0, stream>>>(qgm, qnw, cosT, sinT, qbf, NH_, 2 * D_);
  prep_norm_rope<<<(M * NKV_) / 8, 256, 0, stream>>>(kproj, knw, cosT, sinT, kbf, NKV_, D_);
  prep_v<<<(M * NKV_ * D_) / 1024, 256, 0, stream>>>(vproj, vbf);
  flash_attn<<<dim3(S_ / 128, NH_, B_), 256, 0, stream>>>(qbf, kbf, vbf, qgm, outp);
}
